// Head_30597347017102
// MI455X (gfx1250) — matrix operand order repaired and checked by reading, not compiled, not run on hardware
//
#include <hip/hip_runtime.h>

typedef __attribute__((ext_vector_type(16))) _Float16 v16h;
typedef __attribute__((ext_vector_type(8)))  _Float16 v8h;
typedef __attribute__((ext_vector_type(8)))  float    v8f;
typedef __attribute__((ext_vector_type(4)))  float    f32x4;

#define BB 64
#define TT 512
#define CC 384
#define HH 64

// ---------------------------------------------------------------------------
// Phase 1: fused QKV projection. One block = 16 rows of x; 3 waves, one per
// output matrix (q/k/v). Each wave computes the full 16x64 tile (4 N-tiles)
// from a single shared A-fragment load per k-step: 12 k-steps x 4 wmma.
// q/k stored row-major fp16 [B,T,H]; v stored TRANSPOSED fp16 [B,H,T] so the
// attention kernel's P@V B-fragments become single contiguous v16h loads.
// grid: (B*T/16), block: (32,3)
// ---------------------------------------------------------------------------
__global__ __launch_bounds__(96) void qkv_proj_kernel(
    const float* __restrict__ x,
    const float* __restrict__ Wq, const float* __restrict__ bq,
    const float* __restrict__ Wk, const float* __restrict__ bk,
    const float* __restrict__ Wv, const float* __restrict__ bv,
    _Float16* __restrict__ qf, _Float16* __restrict__ kf, _Float16* __restrict__ vtf)
{
    const int lane = threadIdx.x;       // 0..31 (one wave per threadIdx.y)
    const int mat  = threadIdx.y;       // 0=q, 1=k, 2=v
    const int m0   = blockIdx.x * 16;   // row tile within [0, B*T)

    const float* W; const float* bias;
    if (mat == 0)      { W = Wq; bias = bq; }
    else if (mat == 1) { W = Wk; bias = bk; }
    else               { W = Wv; bias = bv; }

    const int m     = lane & 15;    // A-matrix row
    const int g     = lane >> 4;    // half-wave group
    const int kbase = g * 16;       // A K-offset per group
    const int n     = lane & 15;    // B/D column within an N-tile

    v8f acc[4] = {v8f{}, v8f{}, v8f{}, v8f{}};
    for (int k0 = 0; k0 < CC; k0 += 32) {
        // ---- A fragment: x tile 16x32 (f32 -> f16), loaded once per k-step ----
        const float* arow = x + (size_t)(m0 + m) * CC + k0 + kbase;
        float af[16];
        *(f32x4*)&af[0]  = *(const f32x4*)(arow);
        *(f32x4*)&af[4]  = *(const f32x4*)(arow + 4);
        *(f32x4*)&af[8]  = *(const f32x4*)(arow + 8);
        *(f32x4*)&af[12] = *(const f32x4*)(arow + 12);
        v16h a;
        #pragma unroll
        for (int j = 0; j < 16; ++j) a[j] = (_Float16)af[j];

        // ---- 4 N-tiles share the A fragment ----
        #pragma unroll
        for (int nt = 0; nt < 4; ++nt) {
            const float* wcol = W + (size_t)(k0 + g * 16) * HH + nt * 16 + n;
            v16h bm;
            #pragma unroll
            for (int j = 0; j < 16; ++j) bm[j] = (_Float16)wcol[(size_t)j * HH];
            acc[nt] = __builtin_amdgcn_wmma_f32_16x16x32_f16(
                          false, a, false, bm, (short)0, acc[nt], false, false);
        }
    }

    if (mat < 2) {
        _Float16* out = (mat == 0) ? qf : kf;
        #pragma unroll
        for (int nt = 0; nt < 4; ++nt) {
            const float bb = bias[nt * 16 + n];
            #pragma unroll
            for (int r = 0; r < 8; ++r) {
                const int row = m0 + 8 * g + r;           // D row = 8*g + r
                out[(size_t)row * HH + nt * 16 + n] = (_Float16)(acc[nt][r] + bb);
            }
        }
    } else {
        // v transposed: vtf[b][h][t]; per lane 8 contiguous t -> one b128 store
        const int bidx = m0 / TT;
        const int t0   = (m0 % TT) + 8 * g;
        #pragma unroll
        for (int nt = 0; nt < 4; ++nt) {
            const float bb = bias[nt * 16 + n];
            v8h pk;
            #pragma unroll
            for (int r = 0; r < 8; ++r) pk[r] = (_Float16)(acc[nt][r] + bb);
            *(v8h*)(vtf + ((size_t)bidx * HH + nt * 16 + n) * TT + t0) = pk;
        }
    }
}

// ---------------------------------------------------------------------------
// Phase 2: causal attention for one (batch, 16-query-row) tile per wave.
//   S = (Q K^T) * H^-0.5 into LDS (fp32), causal-masked
//   softmax stats via LDS + shfl reductions; P = exp(S - max) as fp16 in LDS
//   O = (P V) * (1/rowsum), fp32 out. V is pre-transposed [B,H,T] so every
//   WMMA operand load (Q, K, P, V) is a contiguous 16/32-byte vector load.
// grid: (T/16, B)
// ---------------------------------------------------------------------------
__global__ __launch_bounds__(32) void attn_kernel(
    const _Float16* __restrict__ qf, const _Float16* __restrict__ kf,
    const _Float16* __restrict__ vtf, float* __restrict__ outp)
{
    __shared__ float    Ssc[16][TT];    // 32 KB score tile
    __shared__ _Float16 Ph[16][TT];     // 16 KB exp(S - max) tile
    __shared__ float    rinv_s[16];

    const int lane = threadIdx.x;
    const int q0   = blockIdx.x * 16;
    const int b    = blockIdx.y;
    const size_t base  = (size_t)b * TT * HH;   // for row-major q/k and output
    const size_t basev = (size_t)b * HH * TT;   // for transposed v

    const int n     = lane & 15;
    const int g     = lane >> 4;
    const int kbase = g * 16;

    const int nkt   = (q0 + 16 + 31) / 32;  // 32-wide key tiles needed (causal)
    const int ncols = nkt * 32;

    // ---- load Q tile once as A fragments for k0 = 0 and 32 (H = 64) ----
    v16h aq[2];
    {
        const _Float16* qrow = qf + base + (size_t)(q0 + n) * HH;  // A row = lane&15
        #pragma unroll
        for (int kk = 0; kk < 2; ++kk) {
            const int k0 = kk * 32;
            v8h lo = *(const v8h*)(qrow + k0 + kbase);
            v8h hi = *(const v8h*)(qrow + k0 + 8 + kbase);
            #pragma unroll
            for (int j = 0; j < 8; ++j) { aq[kk][j] = lo[j]; aq[kk][8 + j] = hi[j]; }
        }
    }

    // ---- pass A: scores into LDS ----
    for (int kt = 0; kt < nkt; ++kt) {
        const int s0 = kt * 32;
        #pragma unroll
        for (int sub = 0; sub < 2; ++sub) {
            const int sb = s0 + sub * 16;
            v8f c = {};
            #pragma unroll
            for (int kk = 0; kk < 2; ++kk) {
                // B fragment: rows = h (K-dim), cols = key index
                const _Float16* krow = kf + base + (size_t)(sb + n) * HH + kk * 32 + g * 16;
                v16h bm = *(const v16h*)krow;   // 16 contiguous halves along h
                c = __builtin_amdgcn_wmma_f32_16x16x32_f16(
                        false, aq[kk], false, bm, (short)0, c, false, false);
            }
            #pragma unroll
            for (int r = 0; r < 8; ++r) {
                const int qi = q0 + 8 * g + r;
                const int ki = sb + n;
                float v = c[r] * 0.125f;        // H^-0.5 = 1/8
                if (ki > qi) v = -1.0e30f;      // causal mask
                Ssc[8 * g + r][ki] = v;
            }
        }
    }
    __syncthreads();

    // ---- pass B: per-row max / sum; store unnormalized exp as fp16 ----
    {
        const int row  = lane >> 1;
        const int part = lane & 1;
        float mx = -1.0e30f;
        for (int j = part; j < ncols; j += 2) mx = fmaxf(mx, Ssc[row][j]);
        mx = fmaxf(mx, __shfl_xor(mx, 1, 32));
        float sum = 0.0f;
        for (int j = part; j < ncols; j += 2) {
            const float e = __expf(Ssc[row][j] - mx);
            sum += e;
            Ph[row][j] = (_Float16)e;
        }
        sum += __shfl_xor(sum, 1, 32);
        if (part == 0) rinv_s[row] = 1.0f / sum;  // sum >= 1 (diagonal)
    }
    __syncthreads();

    // ---- pass C: O = P @ V, K-tiles of 32 over LDS-resident P ----
    v8f o[4] = {v8f{}, v8f{}, v8f{}, v8f{}};
    for (int kt = 0; kt < nkt; ++kt) {
        const int s0 = kt * 32;
        // A fragment from Ph: row = lane&15, K contiguous 0..15 (+kbase)
        v16h pa;
        {
            const _Float16* prow = &Ph[n][0];
            v8h lo = *(const v8h*)(prow + s0 + kbase);
            v8h hi = *(const v8h*)(prow + s0 + 8 + kbase);
            #pragma unroll
            for (int j = 0; j < 8; ++j) { pa[j] = lo[j]; pa[8 + j] = hi[j]; }
        }
        #pragma unroll
        for (int nt = 0; nt < 4; ++nt) {
            // B fragment from transposed V: 16 contiguous halves along t (K-dim)
            const _Float16* vrow = vtf + basev + (size_t)(nt * 16 + n) * TT + s0 + g * 16;
            v16h bm = *(const v16h*)vrow;
            o[nt] = __builtin_amdgcn_wmma_f32_16x16x32_f16(
                        false, pa, false, bm, (short)0, o[nt], false, false);
        }
    }

    // ---- epilogue: scale by 1/rowsum, fp32 store ----
    #pragma unroll
    for (int r = 0; r < 8; ++r) {
        const int mrow = 8 * g + r;
        const float rv = rinv_s[mrow];
        float* orow = outp + base + (size_t)(q0 + mrow) * HH;
        #pragma unroll
        for (int nt = 0; nt < 4; ++nt)
            orow[nt * 16 + n] = o[nt][r] * rv;
    }
}

// ---------------------------------------------------------------------------
extern "C" void kernel_launch(void* const* d_in, const int* in_sizes, int n_in,
                              void* d_out, int out_size, void* d_ws, size_t ws_size,
                              hipStream_t stream) {
    const float* x  = (const float*)d_in[0];
    const float* Wq = (const float*)d_in[1];
    const float* bq = (const float*)d_in[2];
    const float* Wk = (const float*)d_in[3];
    const float* bk = (const float*)d_in[4];
    const float* Wv = (const float*)d_in[5];
    const float* bv = (const float*)d_in[6];
    float* out = (float*)d_out;

    // fp16 staging in workspace: q,k row-major [B,T,H]; v transposed [B,H,T]
    _Float16* qf  = (_Float16*)d_ws;
    _Float16* kf  = qf + (size_t)BB * TT * HH;
    _Float16* vtf = kf + (size_t)BB * TT * HH;

    dim3 g1((BB * TT) / 16, 1, 1);
    dim3 b1(32, 3, 1);
    qkv_proj_kernel<<<g1, b1, 0, stream>>>(x, Wq, bq, Wk, bk, Wv, bv, qf, kf, vtf);

    dim3 g2(TT / 16, BB, 1);
    attn_kernel<<<g2, 32, 0, stream>>>(qf, kf, vtf, out);
}
